// BSCAUnrolled_87917980549471
// MI455X (gfx1250) — matrix operand order repaired and checked by reading, not compiled, not run on hardware
//
#include <hip/hip_runtime.h>

// Problem constants (match reference)
#define B_ 8
#define E_ 512
#define F_ 1024
#define T_ 2048
#define RK 10
#define L_ 5

typedef __attribute__((ext_vector_type(16))) __bf16 v16bf;
typedef __attribute__((ext_vector_type(8)))  float  v8f;

union fragu { v16bf v; uint4 q[2]; };
union bf8u  { __bf16 b[8]; uint4 q; };

__device__ __forceinline__ __bf16 f2bf(float f) {
  union { unsigned short u; __bf16 b; } cv;
  unsigned int x = __float_as_uint(f);
  cv.u = (unsigned short)((x + 0x7FFFu + ((x >> 16) & 1u)) >> 16);  // RNE
  return cv.b;
}

__device__ __forceinline__ float sigm(float x) { return 1.f / (1.f + __expf(-x)); }

// ---------------------------------------------------------------------------
// Prep / init kernels (run once per launch; fully overwrite their outputs)
// ---------------------------------------------------------------------------
__global__ __launch_bounds__(256)
void k_prep_R(const float* __restrict__ R, __bf16* __restrict__ Rbf,
              __bf16* __restrict__ RbfT, __bf16* __restrict__ RsqT) {
  int idx = blockIdx.x * 256 + threadIdx.x;           // < B*E*F
  float r = R[idx];
  Rbf[idx] = f2bf(r);
  int f = idx % F_;
  int be = idx / F_;
  int e = be % E_;
  int b = be / E_;
  size_t tidx = ((size_t)b * F_ + f) * E_ + e;
  RbfT[tidx] = f2bf(r);
  RsqT[tidx] = f2bf(r * r);
}

// Omega -> bf16, stored K-major as [B][T][E] (B-operand panel layout)
__global__ __launch_bounds__(256)
void k_prep_Om(const float* __restrict__ Om, __bf16* __restrict__ OmbfT) {
  int idx = blockIdx.x * 256 + threadIdx.x;           // < B*E*T, idx=(b*E+e)*T+t
  int t = idx % T_;
  int be = idx / T_;
  int e = be % E_;
  int b = be / E_;
  OmbfT[((size_t)b * T_ + t) * E_ + e] = f2bf(Om[idx]);
}

// A0 -> outA (f32 [B][F][T]) and AbfT (bf16 K-major [B][T][F])
__global__ __launch_bounds__(256)
void k_init_A(const float* __restrict__ A0, float* __restrict__ outA,
              __bf16* __restrict__ AbfT) {
  int idx = blockIdx.x * 256 + threadIdx.x;           // < B*F*T, idx=(b*F+f)*T+t
  float a = A0[idx];
  outA[idx] = a;
  int t = idx % T_;
  int bf = idx / T_;
  int f = bf % F_;
  int b = bf / F_;
  AbfT[((size_t)b * T_ + t) * F_ + f] = f2bf(a);
}

__global__ __launch_bounds__(256)
void k_init_PQ(const float* __restrict__ P0, const float* __restrict__ Q0,
               float* __restrict__ outP, float* __restrict__ outQ) {
  int idx = blockIdx.x * 256 + threadIdx.x;           // < B*(E+T)*RK
  const int nP = B_ * E_ * RK;
  if (idx < nP) outP[idx] = P0[idx];
  else          outQ[idx - nP] = Q0[idx - nP];
}

// ---------------------------------------------------------------------------
// Tiled bf16 WMMA GEMM: C[b] = Aop[b] (M x K, row-major) * BopT[b]^T
// where BopT is the K-major [N][K] panel of the logical (K x N) B matrix.
// Tiles move memory->LDS with global_load_async_to_lds_b128 (ASYNCcnt),
// double-buffered; fragments are contiguous ds_load_b128 pairs.
// MODE 0: err = Omega*(Y - C)      MODE 1: plainOut = C
// MODE 2: soft-threshold A-update (writes A f32 and AbfT bf16 K-major)
// ---------------------------------------------------------------------------
template <int MODE>
__global__ __launch_bounds__(256)
void k_gemm(const __bf16* __restrict__ Aop, const __bf16* __restrict__ BopT,
            int M, int N, int K,
            const float* __restrict__ Yp, const float* __restrict__ Omp,
            float* __restrict__ errOut,
            float* __restrict__ plainOut,
            const float* __restrict__ Ascale, float* __restrict__ Aio,
            __bf16* __restrict__ AbfT,
            const float* __restrict__ mu_log, const float* __restrict__ skip, int l) {
  const int tid  = threadIdx.x;
  const int lane = tid & 31;
  const int wave = tid >> 5;
  const int h    = lane >> 4;       // half-wave
  const int ln16 = lane & 15;
  const int wm   = wave & 1;        // wave row (2)
  const int wn   = wave >> 1;       // wave col (4)
  const int bx = blockIdx.x, by = blockIdx.y, b = blockIdx.z;

  const size_t Abase = (size_t)b * M * K;
  const size_t Bbase = (size_t)b * N * K;

  __shared__ __align__(16) __bf16 As[2][128][40];   // M-panel rows x 32 K (+pad)
  __shared__ __align__(16) __bf16 Bs[2][128][40];   // N-panel rows x 32 K (+pad)

  v8f acc[4][2];
  v8f zv;
  #pragma unroll
  for (int i = 0; i < 8; ++i) zv[i] = 0.f;
  #pragma unroll
  for (int mi = 0; mi < 4; ++mi)
    #pragma unroll
    for (int ni = 0; ni < 2; ++ni) acc[mi][ni] = zv;

  // per-thread panel-load assignment: 128 rows x 32 cols, 32B per thread
  const int prow = tid >> 1;            // 0..127
  const int pcol = (tid & 1) * 16;      // 0 / 16 (bf16 elements)
  const __bf16* gA = Aop  + Abase + (size_t)(by * 128 + prow) * K + pcol;
  const __bf16* gB = BopT + Bbase + (size_t)(bx * 128 + prow) * K + pcol;
  // low 32 bits of a generic pointer to LDS == wave-relative LDS byte address
  unsigned ldsA[2] = { (unsigned)(unsigned long long)&As[0][prow][pcol],
                       (unsigned)(unsigned long long)&As[1][prow][pcol] };
  unsigned ldsB[2] = { (unsigned)(unsigned long long)&Bs[0][prow][pcol],
                       (unsigned)(unsigned long long)&Bs[1][prow][pcol] };

  auto issue_tile = [&](int buf, int kk) {
    const __bf16* pa = gA + kk;
    const __bf16* pb = gB + kk;
    asm volatile("global_load_async_to_lds_b128 %0, %1, off\n\t"
                 "global_load_async_to_lds_b128 %0, %1, off offset:16"
                 :: "v"(ldsA[buf]), "v"(pa) : "memory");
    asm volatile("global_load_async_to_lds_b128 %0, %1, off\n\t"
                 "global_load_async_to_lds_b128 %0, %1, off offset:16"
                 :: "v"(ldsB[buf]), "v"(pb) : "memory");
  };

  const int nsteps = K >> 5;
  issue_tile(0, 0);
  for (int s = 0; s < nsteps; ++s) {
    asm volatile("s_wait_asynccnt 0x0" ::: "memory");
    __syncthreads();                       // all waves' async writes visible
    if (s + 1 < nsteps) issue_tile((s + 1) & 1, (s + 1) << 5);
    const int buf = s & 1;

    // A fragments (16x32 bf16, ISA 7.12.2): two contiguous 16B runs per lane
    fragu fa[4];
    #pragma unroll
    for (int mi = 0; mi < 4; ++mi) {
      int r = wm * 64 + mi * 16 + ln16;
      fa[mi].q[0] = *(const uint4*)&As[buf][r][h * 8];
      fa[mi].q[1] = *(const uint4*)&As[buf][r][16 + h * 8];
    }
    // B fragments: K-major panel => two contiguous 16B runs per lane (same k order as A)
    fragu fb[2];
    #pragma unroll
    for (int ni = 0; ni < 2; ++ni) {
      int c = wn * 32 + ni * 16 + ln16;
      fb[ni].q[0] = *(const uint4*)&Bs[buf][c][h * 8];
      fb[ni].q[1] = *(const uint4*)&Bs[buf][c][16 + h * 8];
    }
    #pragma unroll
    for (int mi = 0; mi < 4; ++mi)
      #pragma unroll
      for (int ni = 0; ni < 2; ++ni)
        acc[mi][ni] = __builtin_amdgcn_wmma_f32_16x16x32_bf16(
            false, fa[mi].v, false, fb[ni].v, (short)0, acc[mi][ni], false, false);
  }

  float mu = 0.f, s3 = 0.f;
  if (MODE == 2) {
    mu = __expf(mu_log[l]);
    s3 = sigm(skip[l * 3 + 2]);
  }

  #pragma unroll
  for (int mi = 0; mi < 4; ++mi) {
    #pragma unroll
    for (int ni = 0; ni < 2; ++ni) {
      int row0 = by * 128 + wm * 64 + mi * 16;
      int col0 = bx * 128 + wn * 32 + ni * 16;
      int c = col0 + ln16;
      bf8u bb;
      #pragma unroll
      for (int v = 0; v < 8; ++v) {
        int r = row0 + v + 8 * h;     // C/D layout: VGPR v -> M = v + 8*half
        size_t o = (size_t)b * M * N + (size_t)r * N + c;
        float val = acc[mi][ni][v];
        if (MODE == 0) {
          errOut[o] = Omp[o] * (Yp[o] - val);
        } else if (MODE == 1) {
          plainOut[o] = val;
        } else {
          float as = Ascale[o];
          float a  = Aio[o];
          float ba = as * a + val;
          float st = copysignf(fmaxf(fabsf(ba) - mu, 0.f), ba);
          float ah = (as == 0.f) ? 0.f : st / as;
          float an = a + s3 * (ah - a);
          Aio[o] = an;
          bb.b[v] = f2bf(an);
        }
      }
      if (MODE == 2) {
        // 8 contiguous M-values per lane in the K-major bf16 mirror: one 16B store
        *(uint4*)&AbfT[((size_t)b * N + c) * M + row0 + 8 * h] = bb.q;
      }
    }
  }
}

// ---------------------------------------------------------------------------
// wave32 reduction of 65 partials across a 256-thread block
// ---------------------------------------------------------------------------
__device__ __forceinline__ void block_reduce65(float* vals, float* redv, int tid) {
  __shared__ float part[65][8];
  int lane = tid & 31, wave = tid >> 5;
  #pragma unroll
  for (int i = 0; i < 65; ++i) {
    float v = vals[i];
    #pragma unroll
    for (int off = 16; off > 0; off >>= 1) v += __shfl_xor(v, off, 32);
    if (lane == 0) part[i][wave] = v;
  }
  __syncthreads();
  if (tid < 65) {
    float s = 0.f;
    #pragma unroll
    for (int w = 0; w < 8; ++w) s += part[tid][w];
    redv[tid] = s;
  }
  __syncthreads();
}

// SPD 10x10 Cholesky solve; redv[0..9]=rhs, redv[10..64]=lower-tri Gram
__device__ __forceinline__ void chol_solve10(const float* redv, float lam, float* x) {
  float A[10][10];
  int idx = 0;
  for (int i = 0; i < 10; ++i)
    for (int j = 0; j <= i; ++j) { float v = redv[10 + idx++]; A[i][j] = v; A[j][i] = v; }
  for (int i = 0; i < 10; ++i) A[i][i] += lam;
  for (int i = 0; i < 10; ++i)
    for (int j = 0; j <= i; ++j) {
      float sum = A[i][j];
      for (int k = 0; k < j; ++k) sum -= A[i][k] * A[j][k];
      if (i == j) A[i][i] = sqrtf(fmaxf(sum, 1e-30f));
      else        A[i][j] = sum / A[j][j];
    }
  float y[10];
  for (int i = 0; i < 10; ++i) {
    float sum = redv[i];
    for (int k = 0; k < i; ++k) sum -= A[i][k] * y[k];
    y[i] = sum / A[i][i];
  }
  for (int i = 9; i >= 0; --i) {
    float sum = y[i];
    for (int k = i + 1; k < 10; ++k) sum -= A[k][i] * x[k];
    x[i] = sum / A[i][i];
  }
}

// P update: one block per (b,e); Gram over T with old Q
__global__ __launch_bounds__(256)
void k_P_update(const float* __restrict__ Om, const float* __restrict__ err,
                const float* __restrict__ Qm, float* __restrict__ Pm,
                const float* __restrict__ lam_log, const float* __restrict__ skip, int l) {
  int be = blockIdx.x;
  int b = be / E_;
  int tid = threadIdx.x;
  float lam = __expf(lam_log[l]);
  float s = sigm(skip[l * 3 + 0]);
  const float* om = Om + (size_t)be * T_;
  const float* er = err + (size_t)be * T_;
  const float* Qb = Qm + (size_t)b * T_ * RK;
  float vals[65];
  #pragma unroll
  for (int i = 0; i < 65; ++i) vals[i] = 0.f;
  for (int t = tid; t < T_; t += 256) {
    float w = om[t], e2 = er[t];
    float q[RK];
    #pragma unroll
    for (int r = 0; r < RK; ++r) q[r] = Qb[(size_t)t * RK + r];
    #pragma unroll
    for (int i = 0; i < RK; ++i) {
      vals[i] += e2 * q[i];
      #pragma unroll
      for (int j = 0; j <= i; ++j) vals[10 + i * (i + 1) / 2 + j] += w * q[i] * q[j];
    }
  }
  __shared__ float redv[65];
  block_reduce65(vals, redv, tid);
  if (tid == 0) {
    float x[10];
    chol_solve10(redv, lam, x);
    #pragma unroll
    for (int r = 0; r < RK; ++r) {
      float pv = Pm[(size_t)be * RK + r];
      Pm[(size_t)be * RK + r] = pv + s * (x[r] - pv);
    }
  }
}

// Q update: one block per (b,t); Gram over E with new P
__global__ __launch_bounds__(256)
void k_Q_update(const float* __restrict__ Om, const float* __restrict__ err,
                const float* __restrict__ Pm, float* __restrict__ Qm,
                const float* __restrict__ lam_log, const float* __restrict__ skip, int l) {
  int bt = blockIdx.x;
  int b = bt / T_;
  int t = bt % T_;
  int tid = threadIdx.x;
  float lam = __expf(lam_log[l]);
  float s = sigm(skip[l * 3 + 1]);
  float vals[65];
  #pragma unroll
  for (int i = 0; i < 65; ++i) vals[i] = 0.f;
  for (int e = tid; e < E_; e += 256) {
    size_t o = ((size_t)b * E_ + e) * T_ + t;
    float w = Om[o], er = err[o];
    float p[RK];
    #pragma unroll
    for (int r = 0; r < RK; ++r) p[r] = Pm[((size_t)b * E_ + e) * RK + r];
    #pragma unroll
    for (int i = 0; i < RK; ++i) {
      vals[i] += er * p[i];
      #pragma unroll
      for (int j = 0; j <= i; ++j) vals[10 + i * (i + 1) / 2 + j] += w * p[i] * p[j];
    }
  }
  __shared__ float redv[65];
  block_reduce65(vals, redv, tid);
  if (tid == 0) {
    float x[10];
    chol_solve10(redv, lam, x);
    #pragma unroll
    for (int r = 0; r < RK; ++r) {
      float qv = Qm[(size_t)bt * RK + r];
      Qm[(size_t)bt * RK + r] = qv + s * (x[r] - qv);
    }
  }
}

// datafit = err - Omega*(P Q^T), written coalesced into K-major [B][T][E] bf16
__global__ __launch_bounds__(256)
void k_datafit(const float* __restrict__ err, const float* __restrict__ Om,
               const float* __restrict__ Pm, const float* __restrict__ Qm,
               __bf16* __restrict__ dfT) {
  int bt = blockIdx.x;               // (b,t)
  int b = bt / T_;
  int t = bt % T_;
  __shared__ float qs[RK];
  if (threadIdx.x < RK) qs[threadIdx.x] = Qm[(size_t)bt * RK + threadIdx.x];
  __syncthreads();
  float ql[RK];
  #pragma unroll
  for (int r = 0; r < RK; ++r) ql[r] = qs[r];
  for (int e = threadIdx.x; e < E_; e += 256) {
    size_t o = ((size_t)b * E_ + e) * T_ + t;
    float dot = 0.f;
    #pragma unroll
    for (int r = 0; r < RK; ++r) dot += Pm[((size_t)b * E_ + e) * RK + r] * ql[r];
    dfT[((size_t)b * T_ + t) * E_ + e] = f2bf(err[o] - Om[o] * dot);
  }
}

// ---------------------------------------------------------------------------
extern "C" void kernel_launch(void* const* d_in, const int* in_sizes, int n_in,
                              void* d_out, int out_size, void* d_ws, size_t ws_size,
                              hipStream_t stream) {
  (void)in_sizes; (void)n_in; (void)out_size; (void)ws_size;
  const float* Y   = (const float*)d_in[0];
  const float* R   = (const float*)d_in[1];
  const float* Om  = (const float*)d_in[2];
  const float* P0  = (const float*)d_in[3];
  const float* Q0  = (const float*)d_in[4];
  const float* A0  = (const float*)d_in[5];
  const float* lam_log = (const float*)d_in[6];
  const float* mu_log  = (const float*)d_in[7];
  const float* skip    = (const float*)d_in[8];

  float* outP = (float*)d_out;                       // [B,E,RK]
  float* outQ = outP + (size_t)B_ * E_ * RK;         // [B,T,RK]
  float* outA = outQ + (size_t)B_ * T_ * RK;         // [B,F,T]

  char* ws = (char*)d_ws;
  size_t off = 0;
  auto alloc = [&](size_t bytes) -> void* {
    void* p = ws + off;
    off += (bytes + 255) & ~(size_t)255;
    return p;
  };
  __bf16* Rbf    = (__bf16*)alloc((size_t)B_ * E_ * F_ * 2);  // [B,E,F]
  __bf16* RbfT   = (__bf16*)alloc((size_t)B_ * E_ * F_ * 2);  // [B,F,E]
  __bf16* RsqT   = (__bf16*)alloc((size_t)B_ * E_ * F_ * 2);  // [B,F,E]
  __bf16* OmbfT  = (__bf16*)alloc((size_t)B_ * E_ * T_ * 2);  // [B,T,E] K-major
  __bf16* AbfT   = (__bf16*)alloc((size_t)B_ * F_ * T_ * 2);  // [B,T,F] K-major
  float*  err    = (float*) alloc((size_t)B_ * E_ * T_ * 4);  // [B,E,T]
  __bf16* dfbfT  = (__bf16*)alloc((size_t)B_ * E_ * T_ * 2);  // [B,T,E] K-major
  float*  Ascale = (float*) alloc((size_t)B_ * F_ * T_ * 4);  // [B,F,T]

  dim3 blk(256);
  k_prep_R <<<(B_ * E_ * F_) / 256, blk, 0, stream>>>(R, Rbf, RbfT, RsqT);
  k_prep_Om<<<(B_ * E_ * T_) / 256, blk, 0, stream>>>(Om, OmbfT);
  k_init_A <<<(B_ * F_ * T_) / 256, blk, 0, stream>>>(A0, outA, AbfT);
  k_init_PQ<<<(B_ * (E_ + T_) * RK) / 256, blk, 0, stream>>>(P0, Q0, outP, outQ);

  // A_scale = (R*R)^T @ Omega  (loop-invariant, computed once)
  k_gemm<1><<<dim3(T_ / 128, F_ / 128, B_), blk, 0, stream>>>(
      RsqT, OmbfT, F_, T_, E_,
      nullptr, nullptr, nullptr, Ascale, nullptr, nullptr, nullptr, nullptr, nullptr, 0);

  for (int l = 0; l < L_; ++l) {
    // err = Omega * (Y - R @ A)
    k_gemm<0><<<dim3(T_ / 128, E_ / 128, B_), blk, 0, stream>>>(
        Rbf, AbfT, E_, T_, F_,
        Y, Om, err, nullptr, nullptr, nullptr, nullptr, nullptr, nullptr, 0);
    k_P_update<<<B_ * E_, blk, 0, stream>>>(Om, err, outQ, outP, lam_log, skip, l);
    k_Q_update<<<B_ * T_, blk, 0, stream>>>(Om, err, outP, outQ, lam_log, skip, l);
    // datafit = err - Omega*(P Q^T)   (reuses err: avoids a second R@A GEMM)
    k_datafit<<<B_ * T_, blk, 0, stream>>>(err, Om, outP, outQ, dfbfT);
    // G = R^T @ datafit, fused soft-threshold A update
    k_gemm<2><<<dim3(T_ / 128, F_ / 128, B_), blk, 0, stream>>>(
        RbfT, dfbfT, F_, T_, E_,
        nullptr, nullptr, nullptr, nullptr, Ascale, outA, AbfT, mu_log, skip, l);
  }
}
